// SerializedMambaEncoder_11596411699695
// MI455X (gfx1250) — matrix operand order repaired and checked by reading, not compiled, not run on hardware
//
#include <hip/hip_runtime.h>
#include <hip/hip_bf16.h>
#include <math.h>

typedef __attribute__((ext_vector_type(16))) _Float16 v16h;
typedef __attribute__((ext_vector_type(2)))  _Float16 h2;
typedef __attribute__((ext_vector_type(8)))  float    v8f;

#define BSZ   4
#define NPTS  4096
#define DMOD  256
#define DIN   512
#define LSEQ  4098              // N + 2 prompt tokens
#define BL    (BSZ * LSEQ)      // 16392 tokens per scheme
#define BLPAD 16448             // BL padded to multiple of 64
#define PLD   64                // padded proj/W_x column stride

// registers holding one staged 64x32 A chunk + 32x64 B chunk (per thread)
struct StageRegs {
    float2 a[4];
    float  b0[4], b1[4];
};

// ---------------------------------------------------------------------------
// WMMA GEMM: C[M,N] = A[M,K] * B[K,N] (+bias) (+residual), fp32 in/out.
// fp32 via f16 hi/lo split: ah*bh + al*bh + ah*bl (3 WMMAs per tile pair).
// Block = 256 threads = 8 waves; block tile 64x64; wave tile 16x32.
// Software-pipelined: global loads for chunk k+1 prefetched to registers
// while WMMAs consume LDS buffer k; double-buffered LDS, 1 barrier/chunk.
// All dims padded by caller: M % 64 == 0, N % 64 == 0, K % 32 == 0 -> no guards.
// LDS tiles are pre-swizzled into WMMA fragment order (32 contiguous bytes
// per lane) so every fragment is exactly two ds_load_b128.
// EPI: 0 = none, 1 = +bias, 2 = +bias +residual (residual has ldc layout)
// ---------------------------------------------------------------------------
template <int EPI>
__global__ __launch_bounds__(256)
void gemm_wmma(const float* __restrict__ A, const float* __restrict__ Bw,
               const float* __restrict__ bias, const float* __restrict__ resid,
               float* __restrict__ C, int N, int K, int lda, int ldb, int ldc)
{
    // [buf][tile][lane][half-pair] : one lane's 16 halves contiguous (32 B)
    __shared__ h2 AhF[2][4][32][8], AlF[2][4][32][8];
    __shared__ h2 BhF[2][4][32][8], BlF[2][4][32][8];

    const int tid  = threadIdx.x;
    const int row0 = blockIdx.y * 64;
    const int col0 = blockIdx.x * 64;
    const int wave = tid >> 5;
    const int lane = tid & 31;
    const int tA   = wave >> 1;            // A tile (M dir), 0..3
    const int tB0  = (wave & 1) << 1;      // first of two B tiles (N dir)

    // batch all 12 global loads of one chunk back-to-back (single wait)
    auto stage_load = [&](int k0, StageRegs& rg) {
#pragma unroll
        for (int it = 0; it < 4; ++it) {
            int p  = tid + (it << 8);
            int ra = p >> 4, kpa = (p & 15) << 1;
            rg.a[it] = *(const float2*)(A + (size_t)(row0 + ra) * lda + (k0 + kpa));
            int cc = p & 63, kpb = (p >> 6) << 1;
            const float* bp = Bw + (size_t)(k0 + kpb) * ldb + (col0 + cc);
            rg.b0[it] = bp[0];
            rg.b1[it] = bp[ldb];
        }
    };

    // convert to f16 hi/lo pairs and store into fragment-ordered LDS
    auto stage_store = [&](int buf, const StageRegs& rg) {
#pragma unroll
        for (int it = 0; it < 4; ++it) {
            int p = tid + (it << 8);
            {   // A 16x32 f16 layout: lanes0-15 K{0..7,16..23}, lanes16-31 K{8..15,24..31}
                int ra = p >> 4, kp = (p & 15) << 1;
                float2 v = rg.a[it];
                _Float16 a0 = (_Float16)v.x, a1 = (_Float16)v.y;
                h2 hi = { a0, a1 };
                h2 lo = { (_Float16)(v.x - (float)a0), (_Float16)(v.y - (float)a1) };
                int tile   = ra >> 4;
                int lane_w = (ra & 15) + (((kp >> 3) & 1) << 4);
                int i      = (kp & 7) + ((kp >> 4) << 3);      // even
                AhF[buf][tile][lane_w][i >> 1] = hi;
                AlF[buf][tile][lane_w][i >> 1] = lo;
            }
            {   // B 32x16 f16 layout: lanes0-15 K{0..7,16..23}, lanes16-31 K{8..15,24..31}
                int cc = p & 63, kp = (p >> 6) << 1;
                float v0 = rg.b0[it], v1 = rg.b1[it];
                _Float16 b0 = (_Float16)v0, b1 = (_Float16)v1;
                h2 hi = { b0, b1 };
                h2 lo = { (_Float16)(v0 - (float)b0), (_Float16)(v1 - (float)b1) };
                int tile   = cc >> 4;
                int lane_w = (cc & 15) + (((kp >> 3) & 1) << 4);
                int i      = (kp & 7) + ((kp >> 4) << 3);      // even
                BhF[buf][tile][lane_w][i >> 1] = hi;
                BlF[buf][tile][lane_w][i >> 1] = lo;
            }
        }
    };

    v8f acc0 = {}, acc1 = {};
    StageRegs rg;

    stage_load(0, rg);
    stage_store(0, rg);
    __syncthreads();

    const int nk = K >> 5;
    int buf = 0;
    for (int kc = 0; kc < nk; ++kc) {
        const bool more = (kc + 1) < nk;
        if (more) stage_load((kc + 1) << 5, rg);   // prefetch next chunk -> regs

        // fragments: contiguous 32 B per lane -> 2x ds_load_b128 each
        v16h ah  = *(const v16h*)&AhF[buf][tA][lane][0];
        v16h al  = *(const v16h*)&AlF[buf][tA][lane][0];
        v16h bh0 = *(const v16h*)&BhF[buf][tB0][lane][0];
        v16h bl0 = *(const v16h*)&BlF[buf][tB0][lane][0];
        v16h bh1 = *(const v16h*)&BhF[buf][tB0 + 1][lane][0];
        v16h bl1 = *(const v16h*)&BlF[buf][tB0 + 1][lane][0];

        acc0 = __builtin_amdgcn_wmma_f32_16x16x32_f16(false, ah, false, bh0,
                                                      (short)0, acc0, false, false);
        acc0 = __builtin_amdgcn_wmma_f32_16x16x32_f16(false, al, false, bh0,
                                                      (short)0, acc0, false, false);
        acc0 = __builtin_amdgcn_wmma_f32_16x16x32_f16(false, ah, false, bl0,
                                                      (short)0, acc0, false, false);
        acc1 = __builtin_amdgcn_wmma_f32_16x16x32_f16(false, ah, false, bh1,
                                                      (short)0, acc1, false, false);
        acc1 = __builtin_amdgcn_wmma_f32_16x16x32_f16(false, al, false, bh1,
                                                      (short)0, acc1, false, false);
        acc1 = __builtin_amdgcn_wmma_f32_16x16x32_f16(false, ah, false, bl1,
                                                      (short)0, acc1, false, false);

        if (more) stage_store(buf ^ 1, rg);        // fill other LDS buffer
        __syncthreads();
        buf ^= 1;
    }

    // C/D f32 16x16: VGPR r -> M = r (lanes 0-15) or r+8 (lanes 16-31); N = lane&15
    const int lr = lane & 15, kh = lane >> 4;
#pragma unroll
    for (int r = 0; r < 8; ++r) {
        int gr  = row0 + (tA << 4) + r + (kh << 3);
        int gc0 = col0 + (tB0 << 4) + lr;
        int gc1 = gc0 + 16;
        float v0 = acc0[r], v1 = acc1[r];
        if (EPI >= 1) { v0 += bias[gc0]; v1 += bias[gc1]; }
        if (EPI == 2) {
            v0 += resid[(size_t)gr * ldc + gc0];
            v1 += resid[(size_t)gr * ldc + gc1];
        }
        C[(size_t)gr * ldc + gc0] = v0;
        C[(size_t)gr * ldc + gc1] = v1;
    }
}

// ---------------------------------------------------------------------------
// pad W_x (512 x 48) into (512 x 64) zero-padded workspace copy
// ---------------------------------------------------------------------------
__global__ void pad_wx_kernel(const float* __restrict__ Wx, float* __restrict__ Wxp)
{
    int k = blockIdx.x;        // 512
    int c = threadIdx.x;       // 64
    Wxp[k * PLD + c] = (c < 48) ? Wx[k * 48 + c] : 0.f;
}

// ---------------------------------------------------------------------------
// feats = points @ W_in + b_in     (K = 6, trivial)
// ---------------------------------------------------------------------------
__global__ void in_proj_kernel(const float* __restrict__ points,
                               const float* __restrict__ Win,
                               const float* __restrict__ bin,
                               float* __restrict__ feats)
{
    __shared__ float p[6];
    long t = blockIdx.x;                 // token in [0, B*N)
    if (threadIdx.x < 6) p[threadIdx.x] = points[t * 6 + threadIdx.x];
    __syncthreads();
    int d = threadIdx.x;                 // 256
    float acc = bin[d];
#pragma unroll
    for (int i = 0; i < 6; ++i) acc += p[i] * Win[i * DMOD + d];
    feats[t * DMOD + d] = acc;
}

// ---------------------------------------------------------------------------
// Stable bitonic argsort of serialization codes, one block per batch.
// key = (morton_code << 12) | original_index  -> stable, unique keys.
// ---------------------------------------------------------------------------
__global__ __launch_bounds__(1024)
void sort_kernel(const float* __restrict__ points, int* __restrict__ order,
                 int p0, int p1, int p2)
{
    __shared__ int skey[NPTS];
    __shared__ int sidx[NPTS];
    int b = blockIdx.x;

    for (int n = threadIdx.x; n < NPTS; n += 1024) {
        const float* pt = points + ((long)b * NPTS + n) * 6;
        int g[3];
#pragma unroll
        for (int ax = 0; ax < 3; ++ax) {
            int gi = (int)(pt[ax] * 64.f);
            g[ax] = gi < 0 ? 0 : (gi > 63 ? 63 : gi);
        }
        int gp[3] = { g[p0], g[p1], g[p2] };
        int code = 0;
#pragma unroll
        for (int bit = 0; bit < 6; ++bit)
#pragma unroll
            for (int ax = 0; ax < 3; ++ax)
                code |= ((gp[ax] >> bit) & 1) << (3 * bit + (2 - ax));
        skey[n] = (code << 12) | n;
        sidx[n] = n;
    }
    __syncthreads();

    for (int k = 2; k <= NPTS; k <<= 1) {
        for (int j = k >> 1; j > 0; j >>= 1) {
            for (int i = threadIdx.x; i < NPTS; i += 1024) {
                int ixj = i ^ j;
                if (ixj > i) {
                    bool up = ((i & k) == 0);
                    int ka = skey[i], kb = skey[ixj];
                    bool sw = up ? (ka > kb) : (ka < kb);
                    if (sw) {
                        skey[i] = kb; skey[ixj] = ka;
                        int tv = sidx[i]; sidx[i] = sidx[ixj]; sidx[ixj] = tv;
                    }
                }
            }
            __syncthreads();
        }
    }
    for (int n = threadIdx.x; n < NPTS; n += 1024)
        order[b * NPTS + n] = sidx[n];
}

// ---------------------------------------------------------------------------
// seq[b,0]=seq[b,L-1]=prompt; seq[b,l]=feats[b,order[l-1]] otherwise
// ---------------------------------------------------------------------------
__global__ void build_seq_kernel(const float* __restrict__ feats,
                                 const int* __restrict__ order,
                                 const float* __restrict__ emb,
                                 float* __restrict__ seq)
{
    int t = blockIdx.x;                  // [0, B*L)
    int b = t / LSEQ, l = t - b * LSEQ;
    int d = threadIdx.x;                 // 256
    float v;
    if (l == 0 || l == LSEQ - 1) v = emb[d];
    else v = feats[((long)b * NPTS + order[b * NPTS + (l - 1)]) * DMOD + d];
    seq[(long)t * DMOD + d] = v;
}

// ---------------------------------------------------------------------------
// LayerNorm over DIM (one block per row)
// ---------------------------------------------------------------------------
template <int DIM>
__global__ void ln_kernel(const float* __restrict__ x, const float* __restrict__ g,
                          const float* __restrict__ bta, float* __restrict__ out)
{
    __shared__ float red[DIM];
    long t = blockIdx.x;
    int  i = threadIdx.x;
    float v = x[t * DIM + i];
    red[i] = v; __syncthreads();
    for (int s = DIM >> 1; s > 0; s >>= 1) {
        if (i < s) red[i] += red[i + s];
        __syncthreads();
    }
    float m = red[0] * (1.f / DIM);
    __syncthreads();
    float dv = v - m;
    red[i] = dv * dv; __syncthreads();
    for (int s = DIM >> 1; s > 0; s >>= 1) {
        if (i < s) red[i] += red[i + s];
        __syncthreads();
    }
    float var = red[0] * (1.f / DIM);
    out[t * DIM + i] = dv * rsqrtf(var + 1e-5f) * g[i] + bta[i];
}

// ---------------------------------------------------------------------------
// depthwise causal conv (K=4) over xi half of xz, + SiLU
// ---------------------------------------------------------------------------
__global__ __launch_bounds__(DIN)
void conv_silu_kernel(const float* __restrict__ xz, const float* __restrict__ cw,
                      const float* __restrict__ cb, float* __restrict__ xi)
{
    int t = blockIdx.x;                  // [0, B*L)
    int b = t / LSEQ, l = t - b * LSEQ;
    int d = threadIdx.x;                 // 512
    float acc = cb[d];
#pragma unroll
    for (int k = 0; k < 4; ++k) {
        int lp = l - 3 + k;
        if (lp >= 0)
            acc += cw[d * 4 + k] * xz[((long)(b * LSEQ + lp)) * 1024 + d];
    }
    xi[(long)t * DIN + d] = acc / (1.f + expf(-acc));   // silu
}

// ---------------------------------------------------------------------------
// dt = softplus(proj[:, :16] @ Wdt + bdt)      (proj has PLD stride)
// ---------------------------------------------------------------------------
__global__ __launch_bounds__(DIN)
void dt_kernel(const float* __restrict__ proj, const float* __restrict__ Wdt,
               const float* __restrict__ bdt, float* __restrict__ dt)
{
    __shared__ float p[16];
    long t = blockIdx.x;
    if (threadIdx.x < 16) p[threadIdx.x] = proj[t * PLD + threadIdx.x];
    __syncthreads();
    int d = threadIdx.x;
    float acc = bdt[d];
#pragma unroll
    for (int r = 0; r < 16; ++r) acc += p[r] * Wdt[r * DIN + d];
    dt[t * DIN + d] = (acc > 20.f) ? acc : log1pf(expf(acc));
}

// ---------------------------------------------------------------------------
// selective scan, one block per batch, thread per channel d (DS=16 in regs).
// fuses y = (scan_y + D*xi) * silu(z).
// ---------------------------------------------------------------------------
__global__ __launch_bounds__(DIN)
void scan_kernel(const float* __restrict__ proj, const float* __restrict__ dt,
                 const float* __restrict__ xi, const float* __restrict__ xz,
                 const float* __restrict__ A_log, const float* __restrict__ Dskip,
                 float* __restrict__ y)
{
    __shared__ float Bs[32][16];
    __shared__ float Cs[32][16];
    int b = blockIdx.x, d = threadIdx.x;

    float Ad[16], h[16];
#pragma unroll
    for (int s = 0; s < 16; ++s) {
        Ad[s] = -expf(A_log[d * 16 + s]);
        h[s]  = 0.f;
    }
    float dp = Dskip[d];

    for (int l0 = 0; l0 < LSEQ; l0 += 32) {
        int cnt = LSEQ - l0 < 32 ? LSEQ - l0 : 32;
        {   // stage shared B/C rows for this chunk (512 threads -> 1 elem each)
            int r = threadIdx.x >> 4, c = threadIdx.x & 15;
            if (r < cnt) {
                long pr = ((long)(b * LSEQ + l0 + r)) * PLD;
                Bs[r][c] = proj[pr + 16 + c];
                Cs[r][c] = proj[pr + 32 + c];
            } else { Bs[r][c] = 0.f; Cs[r][c] = 0.f; }
        }
        __syncthreads();

        for (int r = 0; r < cnt; ++r) {
            long idx  = (long)(b * LSEQ + l0 + r);
            float dtv = dt[idx * DIN + d];
            float xv  = xi[idx * DIN + d];
            float zv  = xz[idx * 1024 + DIN + d];
            float dtx = dtv * xv;
            float yv  = 0.f;
#pragma unroll
            for (int s = 0; s < 16; ++s) {
                float dA = expf(dtv * Ad[s]);
                h[s] = dA * h[s] + dtx * Bs[r][s];
                yv  += h[s] * Cs[r][s];
            }
            float sz = zv / (1.f + expf(-zv));
            y[idx * DIN + d] = (yv + dp * xv) * sz;
        }
        __syncthreads();
    }
}

// ---------------------------------------------------------------------------
// scatter block output (dropping prompts) back through the permutation
// ---------------------------------------------------------------------------
__global__ void scatter_kernel(const float* __restrict__ blk,
                               const int* __restrict__ order,
                               float* __restrict__ feats)
{
    int t = blockIdx.x;                  // [0, B*N)
    int b = t >> 12, i = t & (NPTS - 1);
    int d = threadIdx.x;
    feats[((long)b * NPTS + order[b * NPTS + i]) * DMOD + d] =
        blk[((long)(b * LSEQ + i + 1)) * DMOD + d];
}

// ---------------------------------------------------------------------------
// pooled = concat(max_n feats, mean_n feats)   (one block per batch)
// ---------------------------------------------------------------------------
__global__ void pool_kernel(const float* __restrict__ feats, float* __restrict__ pooled)
{
    int b = blockIdx.x, d = threadIdx.x;     // 256
    float mx = -3.402823466e38f, sm = 0.f;
    for (int n = 0; n < NPTS; ++n) {
        float v = feats[((long)b * NPTS + n) * DMOD + d];
        mx = fmaxf(mx, v); sm += v;
    }
    pooled[b * 512 + d]       = mx;
    pooled[b * 512 + 256 + d] = sm * (1.f / NPTS);
}

// ---------------------------------------------------------------------------
// head dense 512x512 (M=4 -> too small for WMMA), optional tanh-gelu
// ---------------------------------------------------------------------------
template <bool GELU>
__global__ __launch_bounds__(512)
void head_dense(const float* __restrict__ x, const float* __restrict__ W,
                const float* __restrict__ bias, float* __restrict__ out)
{
    __shared__ float xs[512];
    int b = blockIdx.x, g = threadIdx.x;
    xs[g] = x[b * 512 + g];
    __syncthreads();
    float acc = bias[g];
    for (int k = 0; k < 512; ++k) acc += xs[k] * W[k * 512 + g];
    if (GELU) {
        float c = 0.7978845608028654f * (acc + 0.044715f * acc * acc * acc);
        acc = 0.5f * acc * (1.f + tanhf(c));
    }
    out[b * 512 + g] = acc;
}

// ---------------------------------------------------------------------------
extern "C" void kernel_launch(void* const* d_in, const int* in_sizes, int n_in,
                              void* d_out, int out_size, void* d_ws, size_t ws_size,
                              hipStream_t stream)
{
    const float* points   = (const float*)d_in[0];
    const float* W_in     = (const float*)d_in[1];
    const float* b_in     = (const float*)d_in[2];
    const float* sch_emb  = (const float*)d_in[3];
    const float* ln_g     = (const float*)d_in[4];
    const float* ln_b     = (const float*)d_in[5];
    const float* W_inblk  = (const float*)d_in[6];
    const float* b_inblk  = (const float*)d_in[7];
    const float* conv_w   = (const float*)d_in[8];
    const float* conv_b   = (const float*)d_in[9];
    const float* W_x      = (const float*)d_in[10];
    const float* W_dt     = (const float*)d_in[11];
    const float* b_dt     = (const float*)d_in[12];
    const float* A_log    = (const float*)d_in[13];
    const float* D_skip   = (const float*)d_in[14];
    const float* W_outblk = (const float*)d_in[15];
    const float* b_outblk = (const float*)d_in[16];
    const float* ln2_g    = (const float*)d_in[17];
    const float* ln2_b    = (const float*)d_in[18];
    const float* W1       = (const float*)d_in[19];
    const float* b1       = (const float*)d_in[20];
    const float* W2       = (const float*)d_in[21];
    const float* b2       = (const float*)d_in[22];

    // carve workspace (fp32 elements); token-major buffers padded to BLPAD rows
    float* ws = (float*)d_ws;
    size_t off = 0;
    float* feats = ws + off; off += (size_t)BSZ * NPTS * DMOD;
    int*   order = (int*)(ws + off); off += (size_t)BSZ * NPTS;
    float* seq   = ws + off; off += (size_t)BLPAD * DMOD;
    float* hln   = ws + off; off += (size_t)BLPAD * DMOD;
    float* xz    = ws + off; off += (size_t)BLPAD * 1024;
    float* xi    = ws + off; off += (size_t)BLPAD * DIN;
    float* proj  = ws + off; off += (size_t)BLPAD * PLD;
    float* dtb   = ws + off; off += (size_t)BLPAD * DIN;
    float* yb    = ws + off; off += (size_t)BLPAD * DIN;
    float* blk   = ws + off; off += (size_t)BLPAD * DMOD;
    float* Wxp   = ws + off; off += (size_t)DIN * PLD;
    float* pooled= ws + off; off += 4 * 512;
    float* hn    = ws + off; off += 4 * 512;
    float* h1    = ws + off; off += 4 * 512;

    const int perms[3][3] = { {0,1,2}, {1,2,0}, {2,0,1} };
    const int mT64 = BLPAD / 64;   // 257

    in_proj_kernel<<<BSZ * NPTS, DMOD, 0, stream>>>(points, W_in, b_in, feats);

    for (int s = 0; s < 3; ++s) {
        sort_kernel<<<BSZ, 1024, 0, stream>>>(points, order,
                                              perms[s][0], perms[s][1], perms[s][2]);
        build_seq_kernel<<<BL, DMOD, 0, stream>>>(feats, order,
                                                  sch_emb + s * DMOD, seq);
        ln_kernel<DMOD><<<BL, DMOD, 0, stream>>>(seq, ln_g + s * DMOD,
                                                 ln_b + s * DMOD, hln);
        // xz = hln @ W_inblk[s] + b_inblk[s]   (BLPAD x 1024, K=256)
        gemm_wmma<1><<<dim3(1024 / 64, mT64), 256, 0, stream>>>(
            hln, W_inblk + (size_t)s * DMOD * 1024, b_inblk + s * 1024, nullptr,
            xz, 1024, DMOD, DMOD, 1024, 1024);
        conv_silu_kernel<<<BL, DIN, 0, stream>>>(xz, conv_w + s * DIN * 4,
                                                 conv_b + s * DIN, xi);
        // proj = xi @ Wxp  (BLPAD x 64, K=512; cols 48..63 are zero)
        pad_wx_kernel<<<DIN, PLD, 0, stream>>>(W_x + (size_t)s * DIN * 48, Wxp);
        gemm_wmma<0><<<dim3(1, mT64), 256, 0, stream>>>(
            xi, Wxp, nullptr, nullptr, proj, PLD, DIN, DIN, PLD, PLD);
        dt_kernel<<<BL, DIN, 0, stream>>>(proj, W_dt + (size_t)s * 16 * DIN,
                                          b_dt + s * DIN, dtb);
        scan_kernel<<<BSZ, DIN, 0, stream>>>(proj, dtb, xi, xz,
                                             A_log + (size_t)s * DIN * 16,
                                             D_skip + s * DIN, yb);
        // blk = yb @ W_outblk[s] + b_outblk[s] + seq   (residual)
        gemm_wmma<2><<<dim3(DMOD / 64, mT64), 256, 0, stream>>>(
            yb, W_outblk + (size_t)s * DIN * DMOD, b_outblk + s * DMOD, seq,
            blk, DMOD, DIN, DIN, DMOD, DMOD);
        scatter_kernel<<<BSZ * NPTS, DMOD, 0, stream>>>(blk, order, feats);
    }

    pool_kernel<<<BSZ, DMOD, 0, stream>>>(feats, pooled);
    ln_kernel<512><<<BSZ, 512, 0, stream>>>(pooled, ln2_g, ln2_b, hn);
    head_dense<true ><<<BSZ, 512, 0, stream>>>(hn, W1, b1, h1);
    head_dense<false><<<BSZ, 512, 0, stream>>>(h1, W2, b2, (float*)d_out);
}
